// KLAutoEncoder_62010737820105
// MI455X (gfx1250) — compile-verified, matrix operand order checked by reading, not run on hardware
//
#include <hip/hip_runtime.h>
#include <hip/hip_bf16.h>

#define CCH 96      // channels
#define SNB 16      // neighbors
#define GGR 12      // groups
#define EPSBN 1e-5f

typedef __attribute__((ext_vector_type(16))) _Float16 v16h;
typedef __attribute__((ext_vector_type(8)))  _Float16 v8h;
typedef __attribute__((ext_vector_type(8)))  float    v8f;

__device__ __forceinline__ v8f wmma_f16(v16h a, v16h b, v8f c) {
  return __builtin_amdgcn_wmma_f32_16x16x32_f16(
      false, a, false, b, (short)0, c, false, false);
}
__device__ __forceinline__ v16h join16(v8h lo, v8h hi) {
  return __builtin_shufflevector(lo, hi, 0,1,2,3,4,5,6,7,8,9,10,11,12,13,14,15);
}
// fragment whose two 8-half runs are contiguous (packed weights)
__device__ __forceinline__ v16h load_fragP(const _Float16* p) {
  return join16(*(const v8h*)p, *(const v8h*)(p + 8));
}
// fragment from a row-major f16 row: runs at +0 and +16 halfs
__device__ __forceinline__ v16h load_fragA(const _Float16* p) {
  return join16(*(const v8h*)p, *(const v8h*)(p + 16));
}

// ---------------------------------------------------------------------------
__global__ void k_zero(float* p, int n) {
  int i = blockIdx.x * blockDim.x + threadIdx.x;
  if (i < n) p[i] = 0.f;
}

// feat f32 -> f16 (vectorized by 4)
__global__ void k_packf(const float* __restrict__ f, _Float16* __restrict__ o, int total4) {
  int i = blockIdx.x * blockDim.x + threadIdx.x;
  if (i < total4) {
    const float4 v = ((const float4*)f)[i];
    o[4*i+0] = (_Float16)v.x; o[4*i+1] = (_Float16)v.y;
    o[4*i+2] = (_Float16)v.z; o[4*i+3] = (_Float16)v.w;
  }
}

// pack a [96 x od] f32 weight into WMMA B-fragment layout:
// out[((ct*3+kb)*32 + lane)*16 + 2*jj+e] = W[kmap(jj,e,hi)][ct*16+col] (f16)
__global__ void k_packw(const float* __restrict__ W, _Float16* __restrict__ out,
                        int od, int ncols, int nct) {
  int i = blockIdx.x * blockDim.x + threadIdx.x;
  if (i >= nct * 3 * 32 * 16) return;
  const int e    = i & 1;
  const int jj   = (i >> 1) & 7;
  const int lane = (i >> 4) & 31;
  const int kb   = (i >> 9) % 3;
  const int ct   = (i >> 9) / 3;
  const int hi = lane >> 4, col = lane & 15;
  const int kk = kb*32 + ((jj < 4) ? 0 : 16) + hi*8 + 2*(jj & 3) + e;
  const int c  = ct*16 + col;
  out[i] = (c < ncols) ? (_Float16)W[(size_t)kk * od + c] : (_Float16)0.f;
}

// pack pos-MLP weights with BN folded: per fragment element a float4
// {a*w0, a*w1, a*w2, a*bp1 + shift}  (runs AFTER k_fin1)
__global__ void k_packp1(const float* __restrict__ Wp1, const float* __restrict__ bp1,
                         const float* __restrict__ par, float4* __restrict__ o) {
  int i = blockIdx.x * blockDim.x + threadIdx.x;
  if (i >= 3 * 32 * 16) return;
  const int e    = i & 1;
  const int jj   = (i >> 1) & 7;
  const int lane = (i >> 4) & 31;
  const int kb   = i >> 9;
  const int hi = lane >> 4;
  const int kk = kb*32 + ((jj < 4) ? 0 : 16) + hi*8 + 2*(jj & 3) + e;
  const float a = par[4*CCH + kk], b = par[5*CCH + kk];
  o[i] = make_float4(a*Wp1[kk], a*Wp1[CCH + kk], a*Wp1[2*CCH + kk], a*bp1[kk] + b);
}

// ---------------------------------------------------------------------------
// qkv GEMM from packed f16 operands; one wave per 16x16 tile. BN stats for q,k.
__global__ __launch_bounds__(32) void k_qkv(
    const _Float16* __restrict__ featp, const _Float16* __restrict__ wqkvp,
    const float* __restrict__ bq, const float* __restrict__ bk, const float* __restrict__ bv,
    float* __restrict__ qlin, float* __restrict__ klin, float* __restrict__ vlin,
    float* __restrict__ stats, int n)
{
  const int lane = threadIdx.x;
  const int row  = lane & 15;
  const int hi   = lane >> 4;
  const int mt   = blockIdx.x;
  const int jb   = blockIdx.y;          // 0..17
  const int wsel = jb / 6;
  const int ct   = jb % 6;

  const float* bias; float* outp;
  if (wsel == 0)      { bias = bq; outp = qlin; }
  else if (wsel == 1) { bias = bk; outp = klin; }
  else                { bias = bv; outp = vlin; }

  const _Float16* fr = featp + (size_t)(mt*16 + row) * CCH;
  const _Float16* wp = wqkvp + (size_t)wsel * (6*3*512);

  v8f acc = {};
#pragma unroll
  for (int kb = 0; kb < 3; ++kb) {
    v16h a = load_fragA(fr + kb*32 + hi*8);
    v16h b = load_fragP(wp + ((ct*3 + kb)*32 + lane)*16);
    acc = wmma_f16(a, b, acc);
  }
  const int   cc  = ct*16 + row;
  const float bch = bias[cc];
  float s = 0.f, s2 = 0.f;
#pragma unroll
  for (int r = 0; r < 8; ++r) {
    float v = acc[r] + bch;
    outp[(size_t)(mt*16 + r + 8*hi) * CCH + cc] = v;
    s += v; s2 += v * v;
  }
  if (wsel < 2) {
    atomicAdd(&stats[wsel*2*CCH + cc], s);
    atomicAdd(&stats[wsel*2*CCH + CCH + cc], s2);
  }
}

// ---------------------------------------------------------------------------
// stats for BN(pos @ Wp1 + bp1) over N*S rows
__global__ __launch_bounds__(96) void k_pstats(
    const float* __restrict__ coord, const int* __restrict__ ridx,
    const float* __restrict__ Wp1, const float* __restrict__ bp1,
    float* __restrict__ stats, int n)
{
  __shared__ float spos[128][3];
  const int c     = threadIdx.x;
  const int base  = blockIdx.x * 128;
  const int total = n * SNB;
  for (int r = c; r < 128; r += 96) {
    const int id = base + r;
    float px = 0.f, py = 0.f, pz = 0.f;
    if (id < total) {
      const int n0 = id >> 4;
      const int jj = ridx[id];
      px = coord[jj*3+0] - coord[n0*3+0];
      py = coord[jj*3+1] - coord[n0*3+1];
      pz = coord[jj*3+2] - coord[n0*3+2];
    }
    spos[r][0] = px; spos[r][1] = py; spos[r][2] = pz;
  }
  __syncthreads();
  const float w0 = Wp1[c], w1 = Wp1[CCH + c], w2 = Wp1[2*CCH + c], bb = bp1[c];
  float s = 0.f, s2 = 0.f;
  for (int r = 0; r < 128; ++r) {
    if (base + r < total) {
      const float h = spos[r][0]*w0 + spos[r][1]*w1 + spos[r][2]*w2 + bb;
      s += h; s2 += h * h;
    }
  }
  atomicAdd(&stats[4*CCH + c], s);
  atomicAdd(&stats[5*CCH + c], s2);
}

// ---------------------------------------------------------------------------
__global__ __launch_bounds__(96) void k_fin1(
    const float* __restrict__ stats, float* __restrict__ par,
    const float* gq, const float* betaq, const float* gk, const float* betak,
    const float* gp, const float* betap, int n)
{
  const int c = threadIdx.x;
  const float cN  = (float)n;
  const float cNS = (float)n * SNB;
  { float m = stats[c] / cN;
    float v = stats[CCH + c] / cN - m*m;
    float a = gq[c] * rsqrtf(v + EPSBN);
    par[c] = a; par[CCH + c] = betaq[c] - m*a; }
  { float m = stats[2*CCH + c] / cN;
    float v = stats[3*CCH + c] / cN - m*m;
    float a = gk[c] * rsqrtf(v + EPSBN);
    par[2*CCH + c] = a; par[3*CCH + c] = betak[c] - m*a; }
  { float m = stats[4*CCH + c] / cNS;
    float v = stats[5*CCH + c] / cNS - m*m;
    float a = gp[c] * rsqrtf(v + EPSBN);
    par[4*CCH + c] = a; par[5*CCH + c] = betap[c] - m*a; }
}

// ---------------------------------------------------------------------------
// main fused per-point kernel: hp -> peb (WMMA) -> rel/vg -> t = rel@Ww1 (WMMA)
__global__ __launch_bounds__(32) void k_main(
    const float* __restrict__ coord, const int* __restrict__ ridx,
    const float4* __restrict__ p1pk, const _Float16* __restrict__ wp2p,
    const float* __restrict__ bp2, const _Float16* __restrict__ ww1p,
    const float* __restrict__ bw1,
    const float* __restrict__ qlin, const float* __restrict__ klin,
    const float* __restrict__ vlin, const float* __restrict__ par,
    _Float16* __restrict__ vg, float* __restrict__ tbuf,
    float* __restrict__ wstat, int n)
{
  __shared__ int sidx[16];
  __shared__ __align__(16) _Float16 srel[16][CCH];
  const int lane = threadIdx.x;
  const int row  = lane & 15;
  const int hi   = lane >> 4;
  const int pt   = blockIdx.x;

  if (lane < 16) sidx[lane] = ridx[pt*SNB + lane];
  __syncthreads();

  const int   jn = sidx[row];
  const float px = coord[jn*3+0] - coord[pt*3+0];
  const float py = coord[jn*3+1] - coord[pt*3+1];
  const float pz = coord[jn*3+2] - coord[pt*3+2];

  // hp = relu(BN(pos @ Wp1 + bp1)) in A-fragment layout via folded float4 table
  v16h a_hp[3];
#pragma unroll
  for (int kb = 0; kb < 3; ++kb) {
#pragma unroll
    for (int idx = 0; idx < 16; ++idx) {
      const float4 w = p1pk[(kb*32 + lane)*16 + idx];
      a_hp[kb][idx] = (_Float16)fmaxf(px*w.x + py*w.y + pz*w.z + w.w, 0.f);
    }
  }

  // peb = hp @ Wp2 (+bp2), fused with relation_qk and val_g
#pragma unroll
  for (int ct = 0; ct < 6; ++ct) {
    v8f acc = {};
#pragma unroll
    for (int kb = 0; kb < 3; ++kb) {
      v16h b = load_fragP(wp2p + ((ct*3 + kb)*32 + lane)*16);
      acc = wmma_f16(a_hp[kb], b, acc);
    }
    const int   cc  = ct*16 + row;
    const float bpc = bp2[cc];
    const float qb  = fmaxf(par[cc] * qlin[(size_t)pt*CCH + cc] + par[CCH + cc], 0.f);
    const float ak  = par[2*CCH + cc], bk2 = par[3*CCH + cc];
#pragma unroll
    for (int r = 0; r < 8; ++r) {
      const int   sr  = r + 8*hi;
      const int   jg  = sidx[sr];
      const float peb = acc[r] + bpc;
      const float kb_ = fmaxf(ak * klin[(size_t)jg*CCH + cc] + bk2, 0.f);
      const float rel = kb_ - qb + peb;
      const float vv  = vlin[(size_t)jg*CCH + cc] + peb;
      vg[((size_t)pt*SNB + sr)*CCH + cc] = (_Float16)vv;
      srel[sr][cc] = (_Float16)rel;
    }
  }
  __syncthreads();

  // t = rel @ Ww1 : [16 x 96] x [96 x 12(pad16)]
  v8f tacc = {};
#pragma unroll
  for (int kb = 0; kb < 3; ++kb) {
    v16h a = load_fragA(&srel[row][kb*32 + hi*8]);
    v16h b = load_fragP(ww1p + (kb*32 + lane)*16);
    tacc = wmma_f16(a, b, tacc);
  }
  if (row < GGR) {
    const float bw = bw1[row];
    float s = 0.f, s2 = 0.f;
#pragma unroll
    for (int r = 0; r < 8; ++r) {
      const float tv = tacc[r] + bw;
      tbuf[((size_t)pt*SNB + r + 8*hi)*GGR + row] = tv;
      s += tv; s2 += tv * tv;
    }
    atomicAdd(&wstat[row], s);
    atomicAdd(&wstat[GGR + row], s2);
  }
}

// ---------------------------------------------------------------------------
__global__ __launch_bounds__(32) void k_fin2(
    const float* __restrict__ stats, float* __restrict__ par,
    const float* gw, const float* betaw, int n)
{
  const int g = threadIdx.x;
  if (g < GGR) {
    const float cnt = (float)n * SNB;
    const float m = stats[6*CCH + g] / cnt;
    const float v = stats[6*CCH + GGR + g] / cnt - m*m;
    const float a = gw[g] * rsqrtf(v + EPSBN);
    par[6*CCH + g] = a;
    par[6*CCH + GGR + g] = betaw[g] - m*a;
  }
}

// ---------------------------------------------------------------------------
__global__ __launch_bounds__(32) void k_out(
    const float* __restrict__ tbuf, const int* __restrict__ ridx,
    const float* __restrict__ Ww2, const float* __restrict__ bw2,
    const _Float16* __restrict__ vg, const float* __restrict__ par,
    float* __restrict__ out, int n)
{
  __shared__ float su[16][GGR];
  __shared__ float sw[16][GGR];
  __shared__ float smask[16];
  const int lane = threadIdx.x;
  const int pt   = blockIdx.x;

  for (int i = lane; i < 16*GGR; i += 32) {
    const int s = i / GGR, g = i % GGR;
    const float tv = tbuf[((size_t)pt*SNB + s)*GGR + g];
    su[s][g] = fmaxf(par[6*CCH + g]*tv + par[6*CCH + GGR + g], 0.f);
  }
  if (lane < 16) {
    const int j = ridx[pt*SNB + lane] + 1;
    smask[lane] = (j > 0) ? 1.f : ((j < 0) ? -1.f : 0.f);
  }
  __syncthreads();
  for (int i = lane; i < 16*GGR; i += 32) {
    const int s = i / GGR, g2 = i % GGR;
    float acc = bw2[g2];
#pragma unroll
    for (int g = 0; g < GGR; ++g) acc += su[s][g] * Ww2[g*GGR + g2];
    sw[s][g2] = acc;
  }
  __syncthreads();
  if (lane < GGR) {
    float m = -3.4e38f;
    for (int s = 0; s < 16; ++s) m = fmaxf(m, sw[s][lane]);
    float e[16]; float sum = 0.f;
    for (int s = 0; s < 16; ++s) { e[s] = __expf(sw[s][lane] - m); sum += e[s]; }
    const float inv = 1.f / sum;
    for (int s = 0; s < 16; ++s) sw[s][lane] = e[s] * inv * smask[s];
  }
  __syncthreads();
  for (int c = lane; c < CCH; c += 32) {
    const int g = c >> 3;
    float acc = 0.f;
    for (int s = 0; s < 16; ++s)
      acc += (float)vg[((size_t)pt*SNB + s)*CCH + c] * sw[s][g];
    out[(size_t)pt*CCH + c] = acc;
  }
}

// ---------------------------------------------------------------------------
extern "C" void kernel_launch(void* const* d_in, const int* in_sizes, int n_in,
                              void* d_out, int out_size, void* d_ws, size_t ws_size,
                              hipStream_t stream)
{
  const float* feat  = (const float*)d_in[0];
  const float* coord = (const float*)d_in[1];
  const int*   ridx  = (const int*)  d_in[2];
  const float* Wq  = (const float*)d_in[3];  const float* bq    = (const float*)d_in[4];
  const float* gq  = (const float*)d_in[5];  const float* betaq = (const float*)d_in[6];
  const float* Wk  = (const float*)d_in[7];  const float* bk    = (const float*)d_in[8];
  const float* gk  = (const float*)d_in[9];  const float* betak = (const float*)d_in[10];
  const float* Wv  = (const float*)d_in[11]; const float* bv    = (const float*)d_in[12];
  const float* Wp1 = (const float*)d_in[13]; const float* bp1   = (const float*)d_in[14];
  const float* gp  = (const float*)d_in[15]; const float* betap = (const float*)d_in[16];
  const float* Wp2 = (const float*)d_in[17]; const float* bp2   = (const float*)d_in[18];
  const float* Ww1 = (const float*)d_in[19]; const float* bw1   = (const float*)d_in[20];
  const float* gw  = (const float*)d_in[21]; const float* betaw = (const float*)d_in[22];
  const float* Ww2 = (const float*)d_in[23]; const float* bw2   = (const float*)d_in[24];

  const int n = in_sizes[1] / 3;   // N from coord [N,3]

  // workspace layout (all sub-regions 16B aligned)
  float*    qlin  = (float*)d_ws;
  float*    klin  = qlin + (size_t)n * CCH;
  float*    vlin  = klin + (size_t)n * CCH;
  _Float16* vg    = (_Float16*)(vlin + (size_t)n * CCH);
  float*    tbuf  = (float*)(vg + (size_t)n * SNB * CCH);
  float*    stats = tbuf + (size_t)n * SNB * GGR;       // 6C+2G sums
  float*    par   = stats + 6*CCH + 2*GGR;              // 6C+2G folded BN params
  _Float16* featp = (_Float16*)(par + 6*CCH + 2*GGR);   // feat in f16
  _Float16* wqkvp = featp + (size_t)n * CCH;            // 3 x 9216 halfs
  _Float16* wp2p  = wqkvp + 3*9216;                     // 9216 halfs
  _Float16* ww1p  = wp2p + 9216;                        // 1536 halfs
  float4*   p1pk  = (float4*)(ww1p + 1536);             // 1536 float4

  const int nsum = 6*CCH + 2*GGR;
  k_zero<<<(nsum + 255) / 256, 256, 0, stream>>>(stats, nsum);

  // one-time packs (f32 -> f16 fragment layouts)
  const int tf4 = (n * CCH) / 4;
  k_packf<<<(tf4 + 255) / 256, 256, 0, stream>>>(feat, featp, tf4);
  k_packw<<<(9216 + 255) / 256, 256, 0, stream>>>(Wq,  wqkvp,          CCH, CCH, 6);
  k_packw<<<(9216 + 255) / 256, 256, 0, stream>>>(Wk,  wqkvp + 9216,   CCH, CCH, 6);
  k_packw<<<(9216 + 255) / 256, 256, 0, stream>>>(Wv,  wqkvp + 2*9216, CCH, CCH, 6);
  k_packw<<<(9216 + 255) / 256, 256, 0, stream>>>(Wp2, wp2p,           CCH, CCH, 6);
  k_packw<<<(1536 + 255) / 256, 256, 0, stream>>>(Ww1, ww1p,           GGR, GGR, 1);

  dim3 g1(n / 16, 18);
  k_qkv<<<g1, 32, 0, stream>>>(featp, wqkvp, bq, bk, bv,
                               qlin, klin, vlin, stats, n);
  k_pstats<<<(n * SNB + 127) / 128, 96, 0, stream>>>(coord, ridx, Wp1, bp1, stats, n);
  k_fin1<<<1, 96, 0, stream>>>(stats, par, gq, betaq, gk, betak, gp, betap, n);
  k_packp1<<<(1536 + 255) / 256, 256, 0, stream>>>(Wp1, bp1, par, p1pk);
  k_main<<<n, 32, 0, stream>>>(coord, ridx, p1pk, wp2p, bp2, ww1p, bw1,
                               qlin, klin, vlin, par, vg, tbuf,
                               stats + 6*CCH, n);
  k_fin2<<<1, 32, 0, stream>>>(stats, par, gw, betaw, n);
  k_out<<<n, 32, 0, stream>>>(tbuf, ridx, Ww2, bw2, vg, par, (float*)d_out, n);
}
